// DRNET_61615600828451
// MI455X (gfx1250) — matrix operand order repaired and checked by reading, not compiled, not run on hardware
//
#include <hip/hip_runtime.h>
#include <hip/hip_bf16.h>

typedef __attribute__((ext_vector_type(16))) _Float16 v16h;
typedef __attribute__((ext_vector_type(8)))  float    v8f;

constexpr int   Bn = 8, Cc = 3, Nn = 4096, Kk = 20, DK = 100;
constexpr int   KP = 128;                 // K padded for 4 x 16x16x32 WMMA chunks
constexpr float EPSv = 1e-5f, SLOPEv = 0.2f;

// ---- orderable-uint key transforms (monotone float <-> uint) ----
__device__ inline unsigned ordkey(float f) {
  unsigned u = __float_as_uint(f);
  return (u & 0x80000000u) ? ~u : (u | 0x80000000u);
}
__device__ inline float invkey(unsigned u) {
  return (u & 0x80000000u) ? __uint_as_float(u ^ 0x80000000u) : __uint_as_float(~u);
}

// ---- K0: zero the atomic accumulators (ws is poisoned, not re-zeroed) ----
__global__ void k0_zero(float* __restrict__ stats2, float* __restrict__ hrowsum) {
  int t = threadIdx.x;
  for (int i = t; i < 128; i += 256) stats2[i] = 0.f;
  for (int i = t; i < Bn * 64; i += 256) hrowsum[i] = 0.f;
}

// ---- K0b: pre-swizzle W_op1 (64x100 f32) into WMMA f16 A-operand layout ----
// one vector of 16 halves per (otile, kchunk, lane); 512 vectors total (16 KB)
__global__ __launch_bounds__(512) void k0b_packW(const float* __restrict__ Wop1,
                                                 _Float16* __restrict__ wpack) {
  int tid = threadIdx.x;               // tid = ((ot*4 + kc)*32 + lane)
  int lane = tid & 31;
  int kc = (tid >> 5) & 3;
  int ot = tid >> 7;
  int l15 = lane & 15;
  int hi = (lane >> 4) & 1;
  int o = ot * 16 + l15;
  const float* Wrow = Wop1 + o * DK;
  _Float16* dst = wpack + (size_t)tid * 16;
#pragma unroll
  for (int i = 0; i < 16; ++i) {
    // contiguous K order (matches B): lanes0-15 K {0..15}, lanes16-31 K {16..31}
    int kA = kc * 32 + hi * 16 + i;
    dst[i] = (kA < DK) ? (_Float16)Wrow[kA] : (_Float16)0.f;
  }
}

// ---- K1: fused pairwise-distance + radix-select top-100 + bitonic sort ----
// one 256-thread workgroup per (b,n) row; 4096 keys live in LDS (16KB)
// writes metric directly as f16 (K padded to 128) in WMMA-B row-major order
__global__ __launch_bounds__(256) void k1_topk(const float* __restrict__ x,
                                               _Float16* __restrict__ metric_h,
                                               int* __restrict__ idxTop) {
  __shared__ unsigned ukey[Nn];
  __shared__ unsigned hist[256];
  __shared__ unsigned skey[256];
  __shared__ int scol[256];
  __shared__ int scnt;
  __shared__ unsigned sprefix;
  __shared__ int srem;

  int b = blockIdx.x / Nn;
  int n = blockIdx.x % Nn;
  int tid = threadIdx.x;
  const float* xb = x + (size_t)b * Cc * Nn;
  float xr0 = xb[0 * Nn + n], xr1 = xb[1 * Nn + n], xr2 = xb[2 * Nn + n];
  float xxr = xr0 * xr0 + xr1 * xr1 + xr2 * xr2;

  for (int j = tid; j < Nn; j += 256) {
    float a0 = xb[0 * Nn + j], a1 = xb[1 * Nn + j], a2 = xb[2 * Nn + j];
    float inner = xr0 * a0 + xr1 * a1 + xr2 * a2;
    float xxj = a0 * a0 + a1 * a1 + a2 * a2;
    float d = 2.f * inner - xxr - xxj;   // = -squared distance
    ukey[j] = ordkey(d);
  }
  if (tid == 0) { scnt = 0; sprefix = 0u; srem = DK; }
  __syncthreads();

  // 4-pass (MSB->LSB) radix select of the 100th-largest key
  for (int p = 3; p >= 0; --p) {
    hist[tid] = 0u;
    __syncthreads();
    unsigned mask_hi = (p == 3) ? 0u : (0xFFFFFFFFu << ((p + 1) * 8));
    unsigned pref = sprefix;
    for (int j = tid; j < Nn; j += 256) {
      unsigned u = ukey[j];
      if ((u & mask_hi) == pref)
        atomicAdd(&hist[(u >> (p * 8)) & 255u], 1u);
    }
    __syncthreads();
    if (tid == 0) {
      int r = srem; unsigned cum = 0; int chosen = 0;
      for (int bin = 255; bin >= 0; --bin) {
        unsigned c = hist[bin];
        if (cum + c >= (unsigned)r) { chosen = bin; srem = r - (int)cum; break; }
        cum += c;
      }
      sprefix = pref | ((unsigned)chosen << (p * 8));
    }
    __syncthreads();
  }
  unsigned kth = sprefix;

  // collect every key >= kth (strictly greater + ties), cap at 256
  for (int j = tid; j < Nn; j += 256) {
    unsigned u = ukey[j];
    if (u >= kth) {
      int pos = atomicAdd(&scnt, 1);
      if (pos < 256) { skey[pos] = u; scol[pos] = j; }
    }
  }
  __syncthreads();
  int cnt = scnt < 256 ? scnt : 256;
  if (tid >= cnt) { skey[tid] = 0u; scol[tid] = 0x7FFFFFFF; }
  __syncthreads();

  // bitonic sort 256 entries: key descending, index ascending on ties
  for (int ksz = 2; ksz <= 256; ksz <<= 1) {
    for (int j = ksz >> 1; j > 0; j >>= 1) {
      int i = tid, ixj = i ^ j;
      if (ixj > i) {
        unsigned ka = skey[i], kb2 = skey[ixj];
        int ca = scol[i], cb = scol[ixj];
        bool before = (ka > kb2) || (ka == kb2 && ca < cb);
        bool up = ((i & ksz) == 0);
        if (up != before) {
          skey[i] = kb2; skey[ixj] = ka;
          scol[i] = cb;  scol[ixj] = ca;
        }
      }
      __syncthreads();
    }
  }

  if (tid < KP) {
    _Float16 hv = (_Float16)0.f;
    if (tid < DK) {
      float v = invkey(skey[tid]);
      hv = (_Float16)(-v);                               // metric = -topk vals
      idxTop[((size_t)b * Nn + n) * DK + tid] =
          (scol[tid] == 0x7FFFFFFF) ? 0 : scol[tid];
    }
    metric_h[((size_t)b * Nn + n) * KP + tid] = hv;      // f16, zero-padded K
  }
}

// ---- K2: m = W_op1(64x100) @ metric^T via v_wmma_f32_16x16x32_f16 ----
// operands pre-converted/pre-swizzled: pure load + WMMA inner loop
__global__ __launch_bounds__(32) void k2_gemm(const _Float16* __restrict__ wpack,
                                              const _Float16* __restrict__ metric_h,
                                              float* __restrict__ mraw) {
  int id = blockIdx.x;
  int nt = id % (Nn / 16); id /= (Nn / 16);
  int ot = id % 4;
  int b  = id / 4;
  int lane = threadIdx.x;
  int l15 = lane & 15;
  int hi = (lane >> 4) & 1;
  int nn = nt * 16 + l15;

  const v16h* Ap = (const v16h*)wpack + (size_t)(ot * 4) * 32 + lane;
  const _Float16* Brow = metric_h + ((size_t)b * Nn + nn) * KP + hi * 16;

  v8f acc = {};
#pragma unroll
  for (int kc = 0; kc < 4; ++kc) {
    v16h a  = Ap[kc * 32];                               // 32B aligned vector load
    v16h bb = *(const v16h*)(Brow + kc * 32);            // 32B aligned vector load
    acc = __builtin_amdgcn_wmma_f32_16x16x32_f16(false, a, false, bb,
                                                 (short)0, acc, false, false);
  }
#pragma unroll
  for (int r = 0; r < 8; ++r) {
    int oo = ot * 16 + r + hi * 8;   // C/D layout: VGPR r -> M = r (+8 for hi lanes)
    mraw[((size_t)b * 64 + oo) * Nn + nn] = acc[r];
  }
}

// ---- K3: per-channel mean/invstd of mraw over (B,N) ----
__global__ __launch_bounds__(256) void k3_stats1(const float* __restrict__ mraw,
                                                 float* __restrict__ stats1) {
  __shared__ float ssum[256], ssq[256];
  int o = blockIdx.x, tid = threadIdx.x;
  float s = 0.f, q = 0.f;
  for (int b = 0; b < Bn; ++b) {
    const float* p = mraw + ((size_t)b * 64 + o) * Nn;
    for (int n = tid; n < Nn; n += 256) { float v = p[n]; s += v; q += v * v; }
  }
  ssum[tid] = s; ssq[tid] = q; __syncthreads();
  for (int st = 128; st > 0; st >>= 1) {
    if (tid < st) { ssum[tid] += ssum[tid + st]; ssq[tid] += ssq[tid + st]; }
    __syncthreads();
  }
  if (tid == 0) {
    float cnt = (float)(Bn * Nn);
    float mean = ssum[0] / cnt;
    float var = ssq[0] / cnt - mean * mean;
    stats1[o] = mean;
    stats1[64 + o] = rsqrtf(var + EPSv);
  }
}

// ---- K4: BN+lrelu, 1x64 head, sigmoid-binning, neighbor-index selection ----
__global__ __launch_bounds__(64) void k4_select(const float* __restrict__ mraw,
    const float* __restrict__ stats1, const float* __restrict__ gam,
    const float* __restrict__ bet, const float* __restrict__ W11,
    const float* __restrict__ bias11, const int* __restrict__ idxTop,
    int* __restrict__ selIdx) {
  int gid = blockIdx.x * 64 + threadIdx.x;
  if (gid >= Bn * Nn) return;
  int b = gid / Nn, n = gid % Nn;
  float acc = bias11[0];
  for (int o = 0; o < 64; ++o) {
    float v = mraw[((size_t)b * 64 + o) * Nn + n];
    float y = (v - stats1[o]) * stats1[64 + o] * gam[o] + bet[o];
    y = (y >= 0.f) ? y : SLOPEv * y;
    acc += W11[o] * y;
  }
  float mval = 5.f / (1.f + __expf(acc)) + 0.5f;   // 5*sigmoid(-acc)+0.5
  int value = (int)floorf(mval + 0.5f);
  value = value < 1 ? 1 : (value > 5 ? 5 : value);
  const int* row = idxTop + (size_t)gid * DK;
  int* out = selIdx + (size_t)gid * Kk;
  for (int k = 0; k < Kk; ++k) out[k] = row[k * value];
}

// ---- K5a: edge-conv pre-activation stats (recompute in K6, never stored) ----
__global__ __launch_bounds__(256) void k5a_edgestats(const float* __restrict__ x,
    const int* __restrict__ selIdx, const float* __restrict__ W1,
    float* __restrict__ stats2) {
  __shared__ float sred[256], sred2[256];
  int tid = threadIdx.x;
  int p = tid >> 6, o = tid & 63;
  int gid = blockIdx.x * 4 + p;
  int b = gid / Nn, n = gid % Nn;
  const float* xb = x + (size_t)b * Cc * Nn;
  float c0 = xb[n], c1 = xb[Nn + n], c2 = xb[2 * Nn + n];
  const float* w = W1 + o * 6;
  float w0 = w[0], w1 = w[1], w2 = w[2];
  float ctr_term = w[3] * c0 + w[4] * c1 + w[5] * c2;
  const int* sel = selIdx + (size_t)gid * Kk;
  float s = 0.f, q = 0.f;
  for (int k = 0; k < Kk; ++k) {
    int j = sel[k];
    float d0 = xb[j] - c0, d1 = xb[Nn + j] - c1, d2 = xb[2 * Nn + j] - c2;
    float v = w0 * d0 + w1 * d1 + w2 * d2 + ctr_term;
    s += v; q += v * v;
  }
  sred[tid] = s; sred2[tid] = q; __syncthreads();
  if (tid < 64) {
    float ts = sred[tid] + sred[tid + 64] + sred[tid + 128] + sred[tid + 192];
    float tq = sred2[tid] + sred2[tid + 64] + sred2[tid + 128] + sred2[tid + 192];
    atomicAdd(&stats2[tid], ts);
    atomicAdd(&stats2[64 + tid], tq);
  }
}

// ---- K5s: turn sums into (mean, invstd) in place ----
__global__ void k5s_finalize(float* __restrict__ stats2) {
  int o = threadIdx.x;   // 64 threads
  float cnt = (float)Bn * (float)Nn * (float)Kk;
  float mean = stats2[o] / cnt;
  float var = stats2[64 + o] / cnt - mean * mean;
  stats2[o] = mean;
  stats2[64 + o] = rsqrtf(var + EPSv);
}

// ---- K6: recompute edge conv, BN+lrelu, max over k, row sums for SE ----
__global__ __launch_bounds__(256) void k6_hmax(const float* __restrict__ x,
    const int* __restrict__ selIdx, const float* __restrict__ W1,
    const float* __restrict__ stats2, const float* __restrict__ gam,
    const float* __restrict__ bet, float* __restrict__ h,
    float* __restrict__ hrowsum) {
  __shared__ float sred[256];
  int tid = threadIdx.x;
  int p = tid >> 6, o = tid & 63;
  int gid = blockIdx.x * 4 + p;
  int b = gid / Nn, n = gid % Nn;
  const float* xb = x + (size_t)b * Cc * Nn;
  float c0 = xb[n], c1 = xb[Nn + n], c2 = xb[2 * Nn + n];
  const float* w = W1 + o * 6;
  float w0 = w[0], w1 = w[1], w2 = w[2];
  float ctr_term = w[3] * c0 + w[4] * c1 + w[5] * c2;
  float mean = stats2[o], istd = stats2[64 + o];
  float g = gam[o], be = bet[o];
  const int* sel = selIdx + (size_t)gid * Kk;
  float hm = -3.4e38f;
  for (int k = 0; k < Kk; ++k) {
    int j = sel[k];
    float d0 = xb[j] - c0, d1 = xb[Nn + j] - c1, d2 = xb[2 * Nn + j] - c2;
    float v = w0 * d0 + w1 * d1 + w2 * d2 + ctr_term;
    float y = (v - mean) * istd * g + be;
    y = (y >= 0.f) ? y : SLOPEv * y;
    hm = fmaxf(hm, y);
  }
  h[((size_t)b * 64 + o) * Nn + n] = hm;
  sred[tid] = hm; __syncthreads();
  if (tid < 64) {
    int b0 = (blockIdx.x * 4) / Nn;   // all 4 points share a batch (N % 4 == 0)
    float ts = sred[tid] + sred[tid + 64] + sred[tid + 128] + sred[tid + 192];
    atomicAdd(&hrowsum[b0 * 64 + tid], ts);
  }
}

// ---- K7: squeeze-excite (tiny: B=8) in one workgroup ----
__global__ __launch_bounds__(256) void k7_se(const float* __restrict__ hrowsum,
    const float* __restrict__ Wsq, const float* __restrict__ gsq,
    const float* __restrict__ bsq, const float* __restrict__ Wex,
    const float* __restrict__ gex, const float* __restrict__ bex,
    float* __restrict__ sv) {
  __shared__ float sin_[Bn * 64];
  __shared__ float qv[Bn * 8];
  __shared__ float ev[Bn * 64];
  int tid = threadIdx.x;
  for (int i = tid; i < Bn * 64; i += 256) sin_[i] = hrowsum[i] / (float)Nn;
  __syncthreads();
  if (tid < Bn * 8) {
    int b = tid / 8, j = tid % 8;
    float acc = 0.f;
    for (int o = 0; o < 64; ++o) acc += Wsq[j * 64 + o] * sin_[b * 64 + o];
    qv[tid] = acc;
  }
  __syncthreads();
  if (tid < 8) {
    float m = 0.f, q2 = 0.f;
    for (int b = 0; b < Bn; ++b) { float v = qv[b * 8 + tid]; m += v; q2 += v * v; }
    m /= (float)Bn;
    float is = rsqrtf(q2 / (float)Bn - m * m + EPSv);
    for (int b = 0; b < Bn; ++b) {
      float v = (qv[b * 8 + tid] - m) * is * gsq[tid] + bsq[tid];
      qv[b * 8 + tid] = v > 0.f ? v : 0.f;
    }
  }
  __syncthreads();
  for (int i = tid; i < Bn * 64; i += 256) {
    int b = i / 64, o = i % 64;
    float acc = 0.f;
    for (int j = 0; j < 8; ++j) acc += Wex[o * 8 + j] * qv[b * 8 + j];
    ev[i] = acc;
  }
  __syncthreads();
  if (tid < 64) {
    float m = 0.f, q2 = 0.f;
    for (int b = 0; b < Bn; ++b) { float v = ev[b * 64 + tid]; m += v; q2 += v * v; }
    m /= (float)Bn;
    float is = rsqrtf(q2 / (float)Bn - m * m + EPSv);
    for (int b = 0; b < Bn; ++b) {
      float v = (ev[b * 64 + tid] - m) * is * gex[tid] + bex[tid];
      sv[b * 64 + tid] = 1.f / (1.f + __expf(-v));
    }
  }
}

// ---- K8: out = h * s ----
__global__ __launch_bounds__(256) void k8_out(const float* __restrict__ h,
    const float* __restrict__ sv, float* __restrict__ out) {
  size_t i = (size_t)blockIdx.x * 256 + threadIdx.x;
  if (i >= (size_t)Bn * 64 * Nn) return;
  out[i] = h[i] * sv[i / Nn];
}

extern "C" void kernel_launch(void* const* d_in, const int* in_sizes, int n_in,
                              void* d_out, int out_size, void* d_ws, size_t ws_size,
                              hipStream_t stream) {
  const float* x      = (const float*)d_in[0];
  const float* Wop1   = (const float*)d_in[1];
  const float* g_op1  = (const float*)d_in[2];
  const float* b_op1  = (const float*)d_in[3];
  const float* Wop11  = (const float*)d_in[4];
  const float* b_op11 = (const float*)d_in[5];
  const float* W1     = (const float*)d_in[6];
  const float* g1     = (const float*)d_in[7];
  const float* b1     = (const float*)d_in[8];
  const float* W_sq   = (const float*)d_in[9];
  const float* g_sq   = (const float*)d_in[10];
  const float* b_sq   = (const float*)d_in[11];
  const float* W_ex   = (const float*)d_in[12];
  const float* g_ex   = (const float*)d_in[13];
  const float* b_ex   = (const float*)d_in[14];

  char* ws = (char*)d_ws;
  _Float16* metric_h = (_Float16*)ws; ws += (size_t)Bn * Nn * KP * 2;   // 8.4 MB
  int*      idxTop   = (int*)ws;      ws += (size_t)Bn * Nn * DK * 4;   // 13.1 MB
  _Float16* wpack    = (_Float16*)ws; ws += (size_t)4 * 4 * 32 * 16 * 2; // 16 KB
  float*    mraw     = (float*)ws;    ws += (size_t)Bn * 64 * Nn * 4;   // 8.4 MB
  int*      selIdx   = (int*)ws;      ws += (size_t)Bn * Nn * Kk * 4;   // 2.6 MB
  float*    hbuf     = (float*)ws;    ws += (size_t)Bn * 64 * Nn * 4;   // 8.4 MB
  float*    stats1   = (float*)ws;    ws += 128 * 4;
  float*    stats2   = (float*)ws;    ws += 128 * 4;
  float*    hrow     = (float*)ws;    ws += Bn * 64 * 4;
  float*    svec     = (float*)ws;    ws += Bn * 64 * 4;

  k0_zero<<<1, 256, 0, stream>>>(stats2, hrow);
  k0b_packW<<<1, 512, 0, stream>>>(Wop1, wpack);
  k1_topk<<<Bn * Nn, 256, 0, stream>>>(x, metric_h, idxTop);
  k2_gemm<<<Bn * 4 * (Nn / 16), 32, 0, stream>>>(wpack, metric_h, mraw);
  k3_stats1<<<64, 256, 0, stream>>>(mraw, stats1);
  k4_select<<<(Bn * Nn + 63) / 64, 64, 0, stream>>>(mraw, stats1, g_op1, b_op1,
                                                    Wop11, b_op11, idxTop, selIdx);
  k5a_edgestats<<<Bn * Nn / 4, 256, 0, stream>>>(x, selIdx, W1, stats2);
  k5s_finalize<<<1, 64, 0, stream>>>(stats2);
  k6_hmax<<<Bn * Nn / 4, 256, 0, stream>>>(x, selIdx, W1, stats2, g1, b1, hbuf, hrow);
  k7_se<<<1, 256, 0, stream>>>(hrow, W_sq, g_sq, b_sq, W_ex, g_ex, b_ex, svec);
  k8_out<<<(Bn * 64 * Nn + 255) / 256, 256, 0, stream>>>(hbuf, svec, (float*)d_out);
}
